// WMN_56109452755621
// MI455X (gfx1250) — compile-verified, matrix operand order checked by reading, not run on hardware
//
#include <hip/hip_runtime.h>
#include <hip/hip_bf16.h>
#include <stdint.h>

// ---------------------------------------------------------------------------
// WMN forward for gfx1250 (MI455X).
//  * All matmuls via v_wmma_f32_16x16x32_f16 (f16 operands, f32 accum).
//  * Weights W[out][in] row-major == Bt[N][K] layout for the WMMA B fragment.
//  * 640-step sequential GRU runs as ONE persistent 512-thread workgroup,
//    hidden state double-buffered in LDS; weight fragment loads kept in-loop
//    (L1/L2 hits) via an opaque ZERO OFFSET (not an opaque pointer!) so LICM
//    cannot hoist them while address-space inference still yields
//    global_load_b128 (LOADcnt only, decoupled from LDS DScnt waits).
//  * kt-outer register tiling everywhere to maximize fragment reuse.
// ---------------------------------------------------------------------------

typedef _Float16 h16;
typedef __attribute__((ext_vector_type(16))) _Float16 v16h;
typedef __attribute__((ext_vector_type(8)))  _Float16 v8h;
typedef __attribute__((ext_vector_type(8)))  float    v8f;

#define NB    32
#define TMS   50
#define WLN   20
#define TQ    20
#define EDIM  128
#define HDIM  256
#define G3H   768
#define HEADS 4
#define HOPS  3
#define VOCAB 32000

// ---- helpers --------------------------------------------------------------
// A (row-major [M][K]) and Bt (row-major [N][K]) share the same per-lane
// pattern: row = lane%16; halves [ (lane/16)*8 .. +8 ) and [ 16+(lane/16)*8 ).
__device__ __forceinline__ v16h frag_ld(const h16* tile, int ld) {
  const int lane = threadIdx.x & 31;
  const h16* p = tile + (size_t)(lane & 15) * ld + ((lane >> 4) << 3);
  v8h lo = *(const v8h*)(p);
  v8h hi = *(const v8h*)(p + 16);
  return __builtin_shufflevector(lo, hi, 0,1,2,3,4,5,6,7,8,9,10,11,12,13,14,15);
}

__device__ __forceinline__ v8f wmma16(v16h a, v16h b, v8f c) {
  // 8 args: (neg_a, A, neg_b, B, c_mod, C, reuse_a, reuse_b)
  return __builtin_amdgcn_wmma_f32_16x16x32_f16(false, a, false, b, (short)0, c,
                                                false, false);
}

// Opaque zero: defeats LICM on fragment loads without breaking pointer
// provenance (keeps global_load_b128 instead of flat_load_b128).
__device__ __forceinline__ size_t opqz() {
  uint32_t z = 0;
  asm volatile("" : "+v"(z));
  return (size_t)z;
}

__device__ __forceinline__ float sigmoidf_(float x) {
  return 1.0f / (1.0f + __expf(-x));
}

// ---- utility kernels ------------------------------------------------------
__global__ void k_f32_to_f16(const float* __restrict__ s, h16* __restrict__ d,
                             size_t n) {
  size_t i = (size_t)blockIdx.x * blockDim.x + threadIdx.x;
  if (i < n) d[i] = (h16)s[i];
}

// xg[n][w][slot(64, zero-padded past 50)][e]
__global__ void k_gather_story(const int* __restrict__ stories,
                               const float* __restrict__ emb,
                               h16* __restrict__ xg) {
  size_t i = (size_t)blockIdx.x * blockDim.x + threadIdx.x;
  const size_t total = (size_t)NB * WLN * 64 * EDIM;
  if (i >= total) return;
  int e = (int)(i & (EDIM - 1));
  size_t r = i >> 7;
  int slot = (int)(r & 63); r >>= 6;
  int w = (int)(r % WLN);
  int n = (int)(r / WLN);
  float v = 0.0f;
  if (slot < TMS) {
    int tok = stories[((size_t)n * TMS + slot) * WLN + w];
    v = emb[(size_t)tok * EDIM + e];
  }
  xg[i] = (h16)v;
}

// xq[w][b][e]
__global__ void k_gather_q(const int* __restrict__ questions,
                           const float* __restrict__ emb,
                           h16* __restrict__ xq) {
  size_t i = (size_t)blockIdx.x * blockDim.x + threadIdx.x;
  const size_t total = (size_t)TQ * NB * EDIM;
  if (i >= total) return;
  int e = (int)(i & (EDIM - 1));
  size_t r = i >> 7;
  int b = (int)(r & 31);
  int w = (int)(r >> 5);
  int tok = questions[(size_t)b * TQ + w];
  xq[i] = (h16)emb[(size_t)tok * EDIM + e];
}

// ---- input-module GRU (persistent, 640 dependent steps) -------------------
__global__ void __launch_bounds__(512)
k_gru_story(const h16* __restrict__ xg,    // [NB][WLN][64][E]
            const h16* __restrict__ Wih,   // [768][128]  (Bt)
            const h16* __restrict__ Whh,   // [768][256]  (Bt)
            const float* __restrict__ bih, const float* __restrict__ bhh,
            h16* __restrict__ mem16) {     // [NB][64][H] (rows>=50 zeroed)
  __shared__ h16 hbuf[2][64 * HDIM];       // 64 KB double buffer
  const int tid  = threadIdx.x;
  const int wave = tid >> 5;               // 0..15, owns H cols [16w,16w+16)
  const int lane = tid & 31;
  const int col16 = lane & 15;
  const int rbase = (lane >> 4) * 8;
  const int jc = wave * 16 + col16;        // this lane's hidden column

  for (int i = tid; i < 64 * HDIM; i += 512) {
    hbuf[0][i] = (h16)0; hbuf[1][i] = (h16)0;
  }
  __syncthreads();

  const float b_r  = bih[jc]            + bhh[jc];
  const float b_z  = bih[HDIM + jc]     + bhh[HDIM + jc];
  const float b_ni = bih[2 * HDIM + jc];
  const float b_nh = bhh[2 * HDIM + jc];

  const h16* WRi0 = Wih + (size_t)(0 * HDIM + wave * 16) * EDIM;
  const h16* WZi0 = Wih + (size_t)(1 * HDIM + wave * 16) * EDIM;
  const h16* WNi0 = Wih + (size_t)(2 * HDIM + wave * 16) * EDIM;
  const h16* WRh0 = Whh + (size_t)(0 * HDIM + wave * 16) * HDIM;
  const h16* WZh0 = Whh + (size_t)(1 * HDIM + wave * 16) * HDIM;
  const h16* WNh0 = Whh + (size_t)(2 * HDIM + wave * 16) * HDIM;

  int cur = 0;
  for (int n = 0; n < NB; ++n) {
    for (int t = 0; t < WLN; ++t) {
      const h16* xbase = xg + ((size_t)(n * WLN + t)) * 64 * EDIM;
      const h16* hprev = hbuf[cur];
      h16* hnext = hbuf[cur ^ 1];
      // opaque per-step offset: weight loads stay in-loop, remain GLOBAL
      const size_t z = opqz();
      const h16* WRi = WRi0 + z; const h16* WZi = WZi0 + z;
      const h16* WNi = WNi0 + z;
      const h16* WRh = WRh0 + z; const h16* WZh = WZh0 + z;
      const h16* WNh = WNh0 + z;

      v8f acc[4][4] = {};   // [mt][R, Z, NI, NH]
      // gi = x @ Wih^T : B fragment loaded once, reused by all 4 row tiles
      #pragma unroll
      for (int kt = 0; kt < 4; ++kt) {
        v16h bR = frag_ld(WRi + kt * 32, EDIM);
        v16h bZ = frag_ld(WZi + kt * 32, EDIM);
        v16h bN = frag_ld(WNi + kt * 32, EDIM);
        #pragma unroll
        for (int mt = 0; mt < 4; ++mt) {
          v16h a = frag_ld(xbase + (size_t)mt * 16 * EDIM + kt * 32, EDIM);
          acc[mt][0] = wmma16(a, bR, acc[mt][0]);
          acc[mt][1] = wmma16(a, bZ, acc[mt][1]);
          acc[mt][2] = wmma16(a, bN, acc[mt][2]);
        }
      }
      // gh = h @ Whh^T
      #pragma unroll
      for (int kt = 0; kt < 8; ++kt) {
        v16h bR = frag_ld(WRh + kt * 32, HDIM);
        v16h bZ = frag_ld(WZh + kt * 32, HDIM);
        v16h bN = frag_ld(WNh + kt * 32, HDIM);
        #pragma unroll
        for (int mt = 0; mt < 4; ++mt) {
          v16h a = frag_ld(hprev + (size_t)mt * 16 * HDIM + kt * 32, HDIM);
          acc[mt][0] = wmma16(a, bR, acc[mt][0]);
          acc[mt][1] = wmma16(a, bZ, acc[mt][1]);
          acc[mt][3] = wmma16(a, bN, acc[mt][3]);
        }
      }
      // gates + state update
      #pragma unroll
      for (int mt = 0; mt < 4; ++mt) {
        #pragma unroll
        for (int i = 0; i < 8; ++i) {
          int row = mt * 16 + rbase + i;
          float hold = (float)hprev[(size_t)row * HDIM + jc];
          float r  = sigmoidf_(acc[mt][0][i] + b_r);
          float z2 = sigmoidf_(acc[mt][1][i] + b_z);
          float nn = tanhf(acc[mt][2][i] + b_ni + r * (acc[mt][3][i] + b_nh));
          hnext[(size_t)row * HDIM + jc] = (h16)((1.0f - z2) * nn + z2 * hold);
        }
      }
      __syncthreads();
      cur ^= 1;
    }
    // export hidden after item n (rows >= 50 -> zero pad for attention)
    for (int i = tid; i < 64 * HDIM; i += 512) {
      int row = i / HDIM;
      mem16[(size_t)n * 64 * HDIM + i] = (row < TMS) ? hbuf[cur][i] : (h16)0;
    }
    __syncthreads();
  }
}

// ---- question GRU (batch 32, 20 steps) ------------------------------------
__global__ void __launch_bounds__(512)
k_gru_q(const h16* __restrict__ xq,    // [TQ][32][E]
        const h16* __restrict__ Wih, const h16* __restrict__ Whh,
        const float* __restrict__ bih, const float* __restrict__ bhh,
        h16* __restrict__ q16) {       // [32][H]
  __shared__ h16 hbuf[2][32 * HDIM];   // 32 KB
  const int tid  = threadIdx.x;
  const int wave = tid >> 5;
  const int lane = tid & 31;
  const int col16 = lane & 15;
  const int rbase = (lane >> 4) * 8;
  const int jc = wave * 16 + col16;

  for (int i = tid; i < 32 * HDIM; i += 512) {
    hbuf[0][i] = (h16)0; hbuf[1][i] = (h16)0;
  }
  __syncthreads();

  const float b_r  = bih[jc]            + bhh[jc];
  const float b_z  = bih[HDIM + jc]     + bhh[HDIM + jc];
  const float b_ni = bih[2 * HDIM + jc];
  const float b_nh = bhh[2 * HDIM + jc];

  const h16* WRi0 = Wih + (size_t)(0 * HDIM + wave * 16) * EDIM;
  const h16* WZi0 = Wih + (size_t)(1 * HDIM + wave * 16) * EDIM;
  const h16* WNi0 = Wih + (size_t)(2 * HDIM + wave * 16) * EDIM;
  const h16* WRh0 = Whh + (size_t)(0 * HDIM + wave * 16) * HDIM;
  const h16* WZh0 = Whh + (size_t)(1 * HDIM + wave * 16) * HDIM;
  const h16* WNh0 = Whh + (size_t)(2 * HDIM + wave * 16) * HDIM;

  int cur = 0;
  for (int t = 0; t < TQ; ++t) {
    const h16* xbase = xq + (size_t)t * 32 * EDIM;
    const h16* hprev = hbuf[cur];
    h16* hnext = hbuf[cur ^ 1];
    const size_t z = opqz();
    const h16* WRi = WRi0 + z; const h16* WZi = WZi0 + z;
    const h16* WNi = WNi0 + z;
    const h16* WRh = WRh0 + z; const h16* WZh = WZh0 + z;
    const h16* WNh = WNh0 + z;

    v8f acc[2][4] = {};
    #pragma unroll
    for (int kt = 0; kt < 4; ++kt) {
      v16h bR = frag_ld(WRi + kt * 32, EDIM);
      v16h bZ = frag_ld(WZi + kt * 32, EDIM);
      v16h bN = frag_ld(WNi + kt * 32, EDIM);
      #pragma unroll
      for (int mt = 0; mt < 2; ++mt) {
        v16h a = frag_ld(xbase + (size_t)mt * 16 * EDIM + kt * 32, EDIM);
        acc[mt][0] = wmma16(a, bR, acc[mt][0]);
        acc[mt][1] = wmma16(a, bZ, acc[mt][1]);
        acc[mt][2] = wmma16(a, bN, acc[mt][2]);
      }
    }
    #pragma unroll
    for (int kt = 0; kt < 8; ++kt) {
      v16h bR = frag_ld(WRh + kt * 32, HDIM);
      v16h bZ = frag_ld(WZh + kt * 32, HDIM);
      v16h bN = frag_ld(WNh + kt * 32, HDIM);
      #pragma unroll
      for (int mt = 0; mt < 2; ++mt) {
        v16h a = frag_ld(hprev + (size_t)mt * 16 * HDIM + kt * 32, HDIM);
        acc[mt][0] = wmma16(a, bR, acc[mt][0]);
        acc[mt][1] = wmma16(a, bZ, acc[mt][1]);
        acc[mt][3] = wmma16(a, bN, acc[mt][3]);
      }
    }
    #pragma unroll
    for (int mt = 0; mt < 2; ++mt) {
      #pragma unroll
      for (int i = 0; i < 8; ++i) {
        int row = mt * 16 + rbase + i;
        float hold = (float)hprev[(size_t)row * HDIM + jc];
        float r  = sigmoidf_(acc[mt][0][i] + b_r);
        float z2 = sigmoidf_(acc[mt][1][i] + b_z);
        float nn = tanhf(acc[mt][2][i] + b_ni + r * (acc[mt][3][i] + b_nh));
        hnext[(size_t)row * HDIM + jc] = (h16)((1.0f - z2) * nn + z2 * hold);
      }
    }
    __syncthreads();
    cur ^= 1;
  }
  for (int i = tid; i < 32 * HDIM; i += 512) q16[i] = hbuf[cur][i];
}

// ---- ft MLP: out = relu(relu(x W1^T + b1) W2^T + b2), x [32,256] ----------
__global__ void __launch_bounds__(256)
k_ft(const h16* __restrict__ x, const h16* __restrict__ W1,
     const float* __restrict__ b1, const h16* __restrict__ W2,
     const float* __restrict__ b2, h16* __restrict__ out) {
  __shared__ h16 mid[32 * HDIM];
  const int tid = threadIdx.x, wave = tid >> 5, lane = tid & 31;
  const int col16 = lane & 15, rbase = (lane >> 4) * 8;
  // wave -> nt = wave, wave+8 ; both mt handled with shared B fragment
  #pragma unroll 2
  for (int half = 0; half < 2; ++half) {
    const h16* X  = half ? mid : x;
    const h16* W  = half ? W2 : W1;
    const float* bb = half ? b2 : b1;
    h16* Y = half ? out : mid;
    #pragma unroll 2
    for (int ni = 0; ni < 2; ++ni) {
      int nt = wave + ni * 8;
      v8f acc0 = {}, acc1 = {};
      #pragma unroll
      for (int kt = 0; kt < 8; ++kt) {
        v16h b = frag_ld(W + (size_t)nt * 16 * HDIM + kt * 32, HDIM);
        acc0 = wmma16(frag_ld(X + (size_t)0  * HDIM + kt * 32, HDIM), b, acc0);
        acc1 = wmma16(frag_ld(X + (size_t)16 * HDIM + kt * 32, HDIM), b, acc1);
      }
      int col = nt * 16 + col16; float bv = bb[col];
      #pragma unroll
      for (int i = 0; i < 8; ++i) {
        float v0 = acc0[i] + bv; v0 = v0 > 0.f ? v0 : 0.f;
        float v1 = acc1[i] + bv; v1 = v1 > 0.f ? v1 : 0.f;
        Y[(size_t)(rbase + i) * HDIM + col] = (h16)v0;
        Y[(size_t)(16 + rbase + i) * HDIM + col] = (h16)v1;
      }
    }
    __syncthreads();
  }
}

// ---- attention: per (n,head) proj(tanh)+score+softmax+weighted sum --------
__global__ void __launch_bounds__(256)
k_attn(const h16* __restrict__ mem16,  // [32][64][256]
       const h16* __restrict__ Wm,     // [4][256][256] (this hop)
       const h16* __restrict__ ok16,   // [32][256]
       h16* __restrict__ rep16) {      // [32][1024]
  __shared__ float score[64];
  __shared__ float sinv;
  const int n = blockIdx.x >> 2, hd = blockIdx.x & 3;
  const int tid = threadIdx.x, wave = tid >> 5, lane = tid & 31;
  const int col16 = lane & 15, rbase = (lane >> 4) * 8;
  const h16* A  = mem16 + (size_t)n * 64 * HDIM;
  const h16* Bt = Wm + (size_t)hd * HDIM * HDIM;
  const h16* ok = ok16 + (size_t)n * HDIM;

  for (int i = tid; i < 64; i += 256) score[i] = 0.f;
  __syncthreads();

  // wave -> mt = wave&3, nt in [ (wave>>2)*8, +8 ): A-frag reused 8x per kt
  const int mt = wave & 3;
  const int ntb = (wave >> 2) * 8;
  v8f acc[8] = {};
  for (int kt = 0; kt < 8; ++kt) {
    v16h a = frag_ld(A + (size_t)mt * 16 * HDIM + kt * 32, HDIM);
    #pragma unroll
    for (int j = 0; j < 8; ++j) {
      v16h b = frag_ld(Bt + (size_t)(ntb + j) * 16 * HDIM + kt * 32, HDIM);
      acc[j] = wmma16(a, b, acc[j]);
    }
  }
  #pragma unroll
  for (int j = 0; j < 8; ++j) {
    float o = (float)ok[(ntb + j) * 16 + col16];
    #pragma unroll
    for (int i = 0; i < 8; ++i)
      atomicAdd(&score[mt * 16 + rbase + i], tanhf(acc[j][i]) * o);
  }
  __syncthreads();
  if (tid == 0) {
    float mx = -1e30f;
    for (int t = 0; t < TMS; ++t) mx = fmaxf(mx, score[t]);
    float s = 0.f;
    for (int t = 0; t < TMS; ++t) { float e = __expf(score[t] - mx); score[t] = e; s += e; }
    sinv = 1.0f / s;
  }
  __syncthreads();
  // rep[d] = sum_t softmax(t) * mem[n][t][d]; one thread per d
  float acc2 = 0.f;
  for (int t = 0; t < TMS; ++t) acc2 += score[t] * (float)A[(size_t)t * HDIM + tid];
  rep16[(size_t)n * (HEADS * HDIM) + hd * HDIM + tid] = (h16)(acc2 * sinv);
}

// ---- buf = rep[32,1024] @ Wo^T + b ----------------------------------------
__global__ void __launch_bounds__(256)
k_buf(const h16* __restrict__ rep16, const h16* __restrict__ Wo, // [256][1024]
      const float* __restrict__ Wob, h16* __restrict__ buf16,
      h16* __restrict__ wb16, int hop) {
  const int tid = threadIdx.x, wave = tid >> 5, lane = tid & 31;
  const int col16 = lane & 15, rbase = (lane >> 4) * 8;
  const int K = HEADS * HDIM;
  v8f acc[2][2] = {};   // [mt][nti]; nt = wave, wave+8
  for (int kt = 0; kt < 32; ++kt) {
    v16h a0 = frag_ld(rep16 + (size_t)0  * K + kt * 32, K);
    v16h a1 = frag_ld(rep16 + (size_t)16 * K + kt * 32, K);
    v16h b0 = frag_ld(Wo + (size_t)wave * 16 * K + kt * 32, K);
    v16h b1 = frag_ld(Wo + (size_t)(wave + 8) * 16 * K + kt * 32, K);
    acc[0][0] = wmma16(a0, b0, acc[0][0]);
    acc[0][1] = wmma16(a0, b1, acc[0][1]);
    acc[1][0] = wmma16(a1, b0, acc[1][0]);
    acc[1][1] = wmma16(a1, b1, acc[1][1]);
  }
  #pragma unroll 2
  for (int mt = 0; mt < 2; ++mt) {
    #pragma unroll 2
    for (int ni = 0; ni < 2; ++ni) {
      int col = (wave + ni * 8) * 16 + col16;
      float bb = Wob[col];
      #pragma unroll
      for (int i = 0; i < 8; ++i) {
        int row = mt * 16 + rbase + i;
        float v = acc[mt][ni][i] + bb;
        buf16[(size_t)row * HDIM + col] = (h16)v;
        wb16[((size_t)row * HOPS + hop) * HDIM + col] = (h16)v;
      }
    }
  }
}

// ---- RM: build comb[288][768] = {wb[n][b], wb[n][a], q[n]} ----------------
__global__ void k_comb(const h16* __restrict__ wb16, const h16* __restrict__ q16,
                       h16* __restrict__ comb) {
  size_t i = (size_t)blockIdx.x * blockDim.x + threadIdx.x;
  if (i >= (size_t)NB * 9 * G3H) return;
  int col = (int)(i % G3H);
  int r = (int)(i / G3H);
  int n = r / 9, p = r % 9, a = p / 3, b_ = p % 3;
  h16 v;
  if (col < HDIM)          v = wb16[((size_t)n * HOPS + b_) * HDIM + col];
  else if (col < 2 * HDIM) v = wb16[((size_t)n * HOPS + a) * HDIM + (col - HDIM)];
  else                     v = q16[(size_t)n * HDIM + (col - 2 * HDIM)];
  comb[i] = v;
}

// ---- Y = relu(X[288,768] @ W^T[768,768] + b) ------------------------------
__global__ void __launch_bounds__(256)
k_mlp768(const h16* __restrict__ X, const h16* __restrict__ W,
         const float* __restrict__ b, h16* __restrict__ Y) {
  const int mt = blockIdx.x;                       // 0..17
  const int tid = threadIdx.x, wave = tid >> 5, lane = tid & 31;
  const int col16 = lane & 15, rbase = (lane >> 4) * 8;
  // wave owns nt = wave*6 .. +6; kt-outer so A-frag is reused 6x
  v8f acc[6] = {};
  for (int kt = 0; kt < 24; ++kt) {
    v16h a = frag_ld(X + (size_t)mt * 16 * G3H + kt * 32, G3H);
    #pragma unroll
    for (int j = 0; j < 6; ++j) {
      v16h bf = frag_ld(W + (size_t)(wave * 6 + j) * 16 * G3H + kt * 32, G3H);
      acc[j] = wmma16(a, bf, acc[j]);
    }
  }
  #pragma unroll
  for (int j = 0; j < 6; ++j) {
    int col = (wave * 6 + j) * 16 + col16; float bb = b[col];
    #pragma unroll
    for (int i = 0; i < 8; ++i) {
      float v = acc[j][i] + bb; v = v > 0.f ? v : 0.f;
      Y[(size_t)(mt * 16 + rbase + i) * G3H + col] = (h16)v;
    }
  }
}

// ---- reasoning[n] = sum_p g2[n*9+p] ---------------------------------------
__global__ void k_reduce9(const h16* __restrict__ g2, h16* __restrict__ reas) {
  int n = blockIdx.x;
  for (int col = threadIdx.x; col < G3H; col += 256) {
    float s = 0.f;
    for (int p = 0; p < 9; ++p) s += (float)g2[((size_t)n * 9 + p) * G3H + col];
    reas[(size_t)n * G3H + col] = (h16)s;
  }
}

// ---- out[32,32000] = reasoning @ V_w^T ------------------------------------
__global__ void __launch_bounds__(256)
k_vocab(const h16* __restrict__ reas16, const h16* __restrict__ Vw, // [32000][768]
        float* __restrict__ out) {
  const int tid = threadIdx.x, wave = tid >> 5, lane = tid & 31;
  const int col16 = lane & 15, rbase = (lane >> 4) * 8;
  const int ntg = blockIdx.x * 8 + wave;          // 0..1999
  v8f acc0 = {}, acc1 = {};
  for (int kt = 0; kt < 24; ++kt) {
    v16h b = frag_ld(Vw + (size_t)ntg * 16 * G3H + kt * 32, G3H);
    acc0 = wmma16(frag_ld(reas16 + (size_t)0  * G3H + kt * 32, G3H), b, acc0);
    acc1 = wmma16(frag_ld(reas16 + (size_t)16 * G3H + kt * 32, G3H), b, acc1);
  }
  int col = ntg * 16 + col16;
  #pragma unroll
  for (int i = 0; i < 8; ++i) {
    out[(size_t)(rbase + i) * VOCAB + col] = acc0[i];
    out[(size_t)(16 + rbase + i) * VOCAB + col] = acc1[i];
  }
}

// ---------------------------------------------------------------------------
extern "C" void kernel_launch(void* const* d_in, const int* in_sizes, int n_in,
                              void* d_out, int out_size, void* d_ws, size_t ws_size,
                              hipStream_t stream) {
  const int*   stories   = (const int*)  d_in[0];
  const int*   questions = (const int*)  d_in[1];
  const float* emb       = (const float*)d_in[2];
  const float* q_Wih     = (const float*)d_in[3];
  const float* q_Whh     = (const float*)d_in[4];
  const float* q_bih     = (const float*)d_in[5];
  const float* q_bhh     = (const float*)d_in[6];
  const float* i_Wih     = (const float*)d_in[7];
  const float* i_Whh     = (const float*)d_in[8];
  const float* i_bih     = (const float*)d_in[9];
  const float* i_bhh     = (const float*)d_in[10];
  const float* Wm        = (const float*)d_in[11];
  const float* Wo_w      = (const float*)d_in[12];
  const float* Wo_b      = (const float*)d_in[13];
  const float* ft_W1     = (const float*)d_in[14];
  const float* ft_b1     = (const float*)d_in[15];
  const float* ft_W2     = (const float*)d_in[16];
  const float* ft_b2     = (const float*)d_in[17];
  const float* g_W1      = (const float*)d_in[18];
  const float* g_b1      = (const float*)d_in[19];
  const float* g_W2      = (const float*)d_in[20];
  const float* g_b2      = (const float*)d_in[21];
  const float* V_w       = (const float*)d_in[22];
  float* out = (float*)d_out;

  // workspace carve-up (all f16, 256B aligned; ~67 MB total)
  size_t off = 0;
  auto take = [&](size_t elems) {
    size_t o = off;
    off = (off + elems * sizeof(h16) + 255) & ~(size_t)255;
    return (h16*)((char*)d_ws + o);
  };
  h16* xg     = take((size_t)NB * WLN * 64 * EDIM);
  h16* xq     = take((size_t)TQ * NB * EDIM);
  h16* iWih16 = take((size_t)G3H * EDIM);
  h16* iWhh16 = take((size_t)G3H * HDIM);
  h16* qWih16 = take((size_t)G3H * EDIM);
  h16* qWhh16 = take((size_t)G3H * HDIM);
  h16* Wm16   = take((size_t)HOPS * HEADS * HDIM * HDIM);
  h16* Wo16   = take((size_t)HOPS * HDIM * HEADS * HDIM);
  h16* ftW116 = take((size_t)HDIM * HDIM);
  h16* ftW216 = take((size_t)HDIM * HDIM);
  h16* gW116  = take((size_t)G3H * G3H);
  h16* gW216  = take((size_t)G3H * G3H);
  h16* Vw16   = take((size_t)VOCAB * G3H);
  h16* mem16  = take((size_t)NB * 64 * HDIM);
  h16* q16    = take((size_t)NB * HDIM);
  h16* ok16   = take((size_t)NB * HDIM);
  h16* buf16  = take((size_t)NB * HDIM);
  h16* rep16  = take((size_t)NB * HEADS * HDIM);
  h16* wb16   = take((size_t)NB * HOPS * HDIM);
  h16* comb16 = take((size_t)NB * 9 * G3H);
  h16* g116   = take((size_t)NB * 9 * G3H);
  h16* g216   = take((size_t)NB * 9 * G3H);
  h16* reas16 = take((size_t)NB * G3H);
  (void)ws_size; (void)n_in; (void)in_sizes; (void)out_size;

  auto conv = [&](const float* s, h16* d, size_t n) {
    k_f32_to_f16<<<(unsigned)((n + 255) / 256), 256, 0, stream>>>(s, d, n);
  };
  conv(i_Wih, iWih16, (size_t)G3H * EDIM);
  conv(i_Whh, iWhh16, (size_t)G3H * HDIM);
  conv(q_Wih, qWih16, (size_t)G3H * EDIM);
  conv(q_Whh, qWhh16, (size_t)G3H * HDIM);
  conv(Wm,    Wm16,   (size_t)HOPS * HEADS * HDIM * HDIM);
  conv(Wo_w,  Wo16,   (size_t)HOPS * HDIM * HEADS * HDIM);
  conv(ft_W1, ftW116, (size_t)HDIM * HDIM);
  conv(ft_W2, ftW216, (size_t)HDIM * HDIM);
  conv(g_W1,  gW116,  (size_t)G3H * G3H);
  conv(g_W2,  gW216,  (size_t)G3H * G3H);
  conv(V_w,   Vw16,   (size_t)VOCAB * G3H);

  {
    size_t n = (size_t)NB * WLN * 64 * EDIM;
    k_gather_story<<<(unsigned)((n + 255) / 256), 256, 0, stream>>>(stories, emb, xg);
    size_t m = (size_t)TQ * NB * EDIM;
    k_gather_q<<<(unsigned)((m + 255) / 256), 256, 0, stream>>>(questions, emb, xq);
  }

  // sequential GRUs (persistent single-WGP workgroups)
  k_gru_story<<<1, 512, 0, stream>>>(xg, iWih16, iWhh16, i_bih, i_bhh, mem16);
  k_gru_q<<<1, 512, 0, stream>>>(xq, qWih16, qWhh16, q_bih, q_bhh, q16);

  // o_k = ft(q)
  k_ft<<<1, 256, 0, stream>>>(q16, ftW116, ft_b1, ftW216, ft_b2, ok16);

  for (int hp = 0; hp < HOPS; ++hp) {
    k_attn<<<NB * HEADS, 256, 0, stream>>>(
        mem16, Wm16 + (size_t)hp * HEADS * HDIM * HDIM, ok16, rep16);
    k_buf<<<1, 256, 0, stream>>>(
        rep16, Wo16 + (size_t)hp * HDIM * HEADS * HDIM, Wo_b + (size_t)hp * HDIM,
        buf16, wb16, hp);
    if (hp < HOPS - 1)
      k_ft<<<1, 256, 0, stream>>>(buf16, ftW116, ft_b1, ftW216, ft_b2, ok16);
  }

  {
    size_t n = (size_t)NB * 9 * G3H;
    k_comb<<<(unsigned)((n + 255) / 256), 256, 0, stream>>>(wb16, q16, comb16);
  }
  k_mlp768<<<18, 256, 0, stream>>>(comb16, gW116, g_b1, g116);
  k_mlp768<<<18, 256, 0, stream>>>(g116,   gW216, g_b2, g216);
  k_reduce9<<<NB, 256, 0, stream>>>(g216, reas16);
  k_vocab<<<VOCAB / (16 * 8), 256, 0, stream>>>(reas16, Vw16, out);
}
